// UpProj_Block_19198503813634
// MI455X (gfx1250) — matrix operand order repaired and checked by reading, not compiled, not run on hardware
//
#include <hip/hip_runtime.h>
#include <hip/hip_bf16.h>
#include <math.h>

// ---------------- types ----------------
typedef __attribute__((ext_vector_type(16))) __bf16 bf16x16;
typedef __attribute__((ext_vector_type(8)))  __bf16 bf16x8;
typedef __attribute__((ext_vector_type(8)))  float  f32x8;

union FragAB { bf16x16 v; bf16x8 h[2]; };

static __device__ __forceinline__ unsigned short f2bf(float f) {
  unsigned int u = __float_as_uint(f);
  unsigned int r = u + 0x7FFFu + ((u >> 16) & 1u);   // round-to-nearest-even
  return (unsigned short)(r >> 16);
}

#define LDSST 40   // bf16 units per LDS row: 80 B (16B-aligned, bank-conflict-free)

// ---------------- packing kernels ----------------
// x: [16,512,32,32] f32 NCHW  ->  Xb: [16,32,32,512] bf16 NHWC
__global__ __launch_bounds__(256) void pack_x_kernel(const float* __restrict__ x,
                                                     unsigned short* __restrict__ Xb) {
  int idx = blockIdx.x * 256 + threadIdx.x;        // NHWC linear, 16*32*32*512
  int cin = idx & 511;
  int t   = idx >> 9;
  int w = t & 31, h = (t >> 5) & 31, n = t >> 10;
  Xb[idx] = f2bf(x[(((size_t)n * 512 + cin) * 32 + h) * 32 + w]);
}

// Pack 8 conv weights (zero-padded to 3x3) into Wp[2048][4608], k = (kh*3+kw)*512 + cin
__global__ __launch_bounds__(256) void pack_w_kernel(
    const float* __restrict__ w1, const float* __restrict__ w2,
    const float* __restrict__ w3, const float* __restrict__ w4,
    const float* __restrict__ w5, const float* __restrict__ w6,
    const float* __restrict__ w7, const float* __restrict__ w8,
    unsigned short* __restrict__ Wp) {
  int k = blockIdx.x * 256 + threadIdx.x;          // 0..4607
  if (k >= 4608) return;
  int co = blockIdx.y;                             // 0..2047
  int conv = co >> 8, c = co & 255;
  int tap = k >> 9, cin = k & 511;
  int kh = tap / 3, kw = tap % 3;
  const float* w = (conv == 0) ? w1 : (conv == 1) ? w2 : (conv == 2) ? w3 :
                   (conv == 3) ? w4 : (conv == 4) ? w5 : (conv == 5) ? w6 :
                   (conv == 6) ? w7 : w8;
  int q = conv & 3;
  float v = 0.0f;
  if (q == 0) {                                    // 3x3
    v = w[(((size_t)c * 512 + cin) * 3 + kh) * 3 + kw];
  } else if (q == 1) {                             // 2x3 (kh in {0,1})
    if (kh < 2) v = w[(((size_t)c * 512 + cin) * 2 + kh) * 3 + kw];
  } else if (q == 2) {                             // 3x2 (kw in {0,1})
    if (kw < 2) v = w[(((size_t)c * 512 + cin) * 3 + kh) * 2 + kw];
  } else {                                         // 2x2
    if (kh < 2 && kw < 2) v = w[(((size_t)c * 512 + cin) * 2 + kh) * 2 + kw];
  }
  Wp[(size_t)co * 4608 + k] = f2bf(v);
}

// w9: [256,256,3,3] -> Wp9[256][2304], k = tap*256 + cin
__global__ __launch_bounds__(256) void pack_w9_kernel(const float* __restrict__ w9,
                                                      unsigned short* __restrict__ Wp9) {
  int k = blockIdx.x * 256 + threadIdx.x;          // 0..2303
  if (k >= 2304) return;
  int co = blockIdx.y;                             // 0..255
  int tap = k >> 8, cin = k & 255;
  int kh = tap / 3, kw = tap % 3;
  Wp9[(size_t)co * 2304 + k] = f2bf(w9[(((size_t)co * 256 + cin) * 3 + kh) * 3 + kw]);
}

__global__ __launch_bounds__(256) void pack_bias_kernel(
    const float* b1, const float* b2, const float* b3, const float* b4,
    const float* b5, const float* b6, const float* b7, const float* b8,
    float* __restrict__ Bias) {
  int t = blockIdx.x * 256 + threadIdx.x;          // 0..2047
  int conv = t >> 8, c = t & 255;
  const float* b = (conv == 0) ? b1 : (conv == 1) ? b2 : (conv == 2) ? b3 :
                   (conv == 3) ? b4 : (conv == 4) ? b5 : (conv == 5) ? b6 :
                   (conv == 6) ? b7 : b8;
  Bias[t] = b[c];
}

__global__ __launch_bounds__(256) void zero_stats_kernel(float* __restrict__ stats) {
  for (int i = threadIdx.x; i < 6 * 256; i += 256) stats[i] = 0.0f;
}

// ---------------- shared implicit-GEMM core ----------------
// Block tile: 256(M) x 128(N), BK = 32.  8 waves as 4(M) x 2(N); each wave owns a
// 64x64 tile = 4x4 accumulators of 16x16 -> 16 v_wmma_f32_16x16x32_bf16 per K-step.
// Double-buffered LDS, last iteration peeled.  Register-pressure discipline:
//   - ALL tile staging is drained to LDS *before* the WMMA block; no staging
//     register crosses it (live set ~190 -> allocator has nothing to spill).
//     The extra s_wait_loadcnt 0 costs little: B loads issue right after A loads,
//     and ~5 resident WGs/WGP (60 KB LDS) cover the per-wave load-wait phase.
//   - 32-bit per-lane global offsets against a wave-uniform base (saddr form)
//   - 9-bit im2col validity bitmask per A-row instead of (h,w) pairs
//   - one B fragment live at a time during the WMMA block
template<int HSZ, int CIN, int KTOT>
static __device__ __forceinline__ void gemm_core(
    const unsigned short* __restrict__ Xsrc,
    const unsigned short* __restrict__ Wsrc,
    int bM, int bN,
    unsigned short* __restrict__ As,   // [2][256*LDSST]
    unsigned short* __restrict__ Bs,   // [2][128*LDSST]
    f32x8 (&acc)[4][4]) {
  constexpr int SH  = (HSZ == 64) ? 6 : 5;
  constexpr int KPT = CIN / 32;                    // K-iterations per tap

  const int tid  = threadIdx.x;
  const int lane = tid & 31;
  const int wave = tid >> 5;
  const int wm = wave >> 1;                        // 0..3 -> 64 rows each
  const int wn = wave & 1;                         // 0..1 -> 64 cols each

  // ---- loop-invariant tile-load addressing (32-bit, per-lane) ----
  const int chunk = tid & 3;                       // 8-bf16 chunk within a 32-K row
  unsigned offA[4], maskA[4];                      // element offset + 9-bit tap validity
  int ldsA[4];
#pragma unroll
  for (int p = 0; p < 4; ++p) {
    const int row = p * 64 + (tid >> 2);           // 0..255
    const int m = bM * 256 + row;
    const int n = m >> (2 * SH);
    const int h = (m >> SH) & (HSZ - 1);
    const int w = m & (HSZ - 1);
    offA[p] = (unsigned)((((n * HSZ + h) * HSZ + w) * CIN) + chunk * 8);
    unsigned mk = 0;
#pragma unroll
    for (int tap = 0; tap < 9; ++tap) {
      const int dh = tap / 3 - 1, dw = tap % 3 - 1;
      if ((unsigned)(h + dh) < (unsigned)HSZ && (unsigned)(w + dw) < (unsigned)HSZ)
        mk |= 1u << tap;
    }
    maskA[p] = mk;
    ldsA[p]  = row * LDSST + chunk * 8;
  }
  unsigned offB[2];
  int ldsB[2];
#pragma unroll
  for (int p = 0; p < 2; ++p) {
    const int row = p * 64 + (tid >> 2);           // 0..127
    offB[p] = (unsigned)((bN * 128 + row) * (9 * CIN) + chunk * 8);
    ldsB[p] = row * LDSST + chunk * 8;
  }

  // ---- loop-invariant fragment addressing (ISA VGPR layouts) ----
  const int kbA  = (lane < 16) ? 0 : 8;            // A 16x32: K halves 0-7/16-23 vs 8-15/24-31
  const int kbB  = (lane < 16) ? 0 : 8;            // B 32x16: K halves 0-7/16-23 vs 8-15/24-31
  const int rlow = lane & 15;
  const int faOff0 = (wm * 64 + rlow) * LDSST + kbA;
  const int fbOff0 = (wn * 64 + rlow) * LDSST + kbB;

  uint4 ra[4], rb[2];
  auto loadA = [&](int kt) {
    const int tap  = kt / KPT;
    const int cinB = (kt % KPT) * 32;
    const int dh = tap / 3 - 1, dw = tap % 3 - 1;
    // wave-uniform base: backend keeps it scalar (saddr) + 32-bit lane offset
    const unsigned short* pa = Xsrc + (((long)dh * HSZ + dw) * CIN + cinB);
#pragma unroll
    for (int p = 0; p < 4; ++p) {
      ra[p] = make_uint4(0u, 0u, 0u, 0u);
      if ((maskA[p] >> tap) & 1u)
        ra[p] = *(const uint4*)(pa + offA[p]);
    }
  };
  auto loadB = [&](int kt) {
    const unsigned short* pb = Wsrc + (unsigned)kt * 32;
#pragma unroll
    for (int p = 0; p < 2; ++p)
      rb[p] = *(const uint4*)(pb + offB[p]);
  };
  auto storeA = [&](int buf) {
    unsigned short* a = As + buf * (256 * LDSST);
#pragma unroll
    for (int p = 0; p < 4; ++p) *(uint4*)(a + ldsA[p]) = ra[p];
  };
  auto storeB = [&](int buf) {
    unsigned short* b = Bs + buf * (128 * LDSST);
#pragma unroll
    for (int p = 0; p < 2; ++p) *(uint4*)(b + ldsB[p]) = rb[p];
  };
  auto compute = [&](int buf) {
    const unsigned short* a = As + buf * (256 * LDSST);
    const unsigned short* b = Bs + buf * (128 * LDSST);
    FragAB fa[4];
#pragma unroll
    for (int i = 0; i < 4; ++i) {
      fa[i].h[0] = *(const bf16x8*)(a + faOff0 + i * (16 * LDSST));
      fa[i].h[1] = *(const bf16x8*)(a + faOff0 + i * (16 * LDSST) + 16);
    }
#pragma unroll
    for (int j = 0; j < 4; ++j) {                  // one B fragment live at a time
      FragAB fb;
      fb.h[0] = *(const bf16x8*)(b + fbOff0 + j * (16 * LDSST));
      fb.h[1] = *(const bf16x8*)(b + fbOff0 + j * (16 * LDSST) + 16);
#pragma unroll
      for (int i = 0; i < 4; ++i)
        acc[i][j] = __builtin_amdgcn_wmma_f32_16x16x32_bf16(
            false, fa[i].v, false, fb.v, (short)0, acc[i][j], false, false);
    }
  };

  loadA(0); loadB(0);
  storeA(0); storeB(0);
  __syncthreads();

  for (int kt = 0; kt < KTOT - 1; ++kt) {          // steady state: branch-free body
    loadA(kt + 1);                                 // 4 global loads issued first
    loadB(kt + 1);                                 // 2 more; arrive ~with the A loads
    storeA((kt & 1) ^ 1);                          // drain staging to LDS *before*
    storeB((kt & 1) ^ 1);                          //   the WMMA block
    compute(kt & 1);
    __syncthreads();
  }
  compute((KTOT - 1) & 1);                         // peeled last iteration
}

// ---------------- GEMM1: both branches' 8 sub-convs ----------------
// M = 16*32*32 = 16384, N = 2048, K = 4608
__global__ __launch_bounds__(256) void gemm1_kernel(
    const unsigned short* __restrict__ Xb,   // [16][32][32][512] bf16
    const unsigned short* __restrict__ Wp,   // [2048][4608] bf16
    const float* __restrict__ Bias,          // [2048]
    float* __restrict__ Y1,                  // [16][64][64][256] f32 (branch1, pre-BN)
    float* __restrict__ Y2) {                // branch2
  __shared__ __align__(16) unsigned short As[2 * 256 * LDSST];
  __shared__ __align__(16) unsigned short Bs[2 * 128 * LDSST];

  f32x8 acc[4][4];
#pragma unroll
  for (int i = 0; i < 4; ++i)
#pragma unroll
    for (int j = 0; j < 4; ++j) acc[i][j] = (f32x8)(0.0f);

  const int bM = blockIdx.x, bN = blockIdx.y;
  gemm_core<32, 512, 144>(Xb, Wp, bM, bN, As, Bs, acc);

  const int lane = threadIdx.x & 31;
  const int wave = threadIdx.x >> 5;
  const int wm = wave >> 1, wn = wave & 1;
  const int rlow = lane & 15;
  // co = conv*256 + c ; conv&3: 0->(0,0) 1->(1,0) 2->(0,1) 3->(1,1); conv<4 -> Y1 else Y2
#pragma unroll
  for (int j = 0; j < 4; ++j) {
    const int co   = bN * 128 + wn * 64 + j * 16 + rlow;
    const float bb = Bias[co];
    const int conv = co >> 8, c = co & 255, q = conv & 3;
    const int dr = q & 1, dc = q >> 1;
    float* Y = (conv < 4) ? Y1 : Y2;
#pragma unroll
    for (int i = 0; i < 4; ++i) {
#pragma unroll
      for (int r = 0; r < 8; ++r) {
        const int m = bM * 256 + wm * 64 + i * 16 + r + (lane >> 4) * 8;
        const int n = m >> 10, hw = m & 1023, h = hw >> 5, w = hw & 31;
        Y[((((size_t)n * 64) + 2 * h + dr) * 64 + (2 * w + dc)) * 256 + c] =
            acc[i][j][r] + bb;
      }
    }
  }
}

// ---------------- GEMM2: conv9 on interleaved 64x64 ----------------
// M = 16*64*64 = 65536, N = 256, K = 2304
__global__ __launch_bounds__(256) void gemm2_kernel(
    const unsigned short* __restrict__ X2,   // [16][64][64][256] bf16 (bn+relu branch1)
    const unsigned short* __restrict__ Wp9,  // [256][2304] bf16
    const float* __restrict__ b9,            // [256]
    float* __restrict__ Z) {                 // [16][64][64][256] f32 (pre-BN)
  __shared__ __align__(16) unsigned short As[2 * 256 * LDSST];
  __shared__ __align__(16) unsigned short Bs[2 * 128 * LDSST];

  f32x8 acc[4][4];
#pragma unroll
  for (int i = 0; i < 4; ++i)
#pragma unroll
    for (int j = 0; j < 4; ++j) acc[i][j] = (f32x8)(0.0f);

  const int bM = blockIdx.x, bN = blockIdx.y;
  gemm_core<64, 256, 72>(X2, Wp9, bM, bN, As, Bs, acc);

  const int lane = threadIdx.x & 31;
  const int wave = threadIdx.x >> 5;
  const int wm = wave >> 1, wn = wave & 1;
  const int rlow = lane & 15;
#pragma unroll
  for (int j = 0; j < 4; ++j) {
    const int c  = bN * 128 + wn * 64 + j * 16 + rlow;   // 0..255
    const float bb = b9[c];
#pragma unroll
    for (int i = 0; i < 4; ++i) {
#pragma unroll
      for (int r = 0; r < 8; ++r) {
        const int m = bM * 256 + wm * 64 + i * 16 + r + (lane >> 4) * 8;
        Z[(size_t)m * 256 + c] = acc[i][j][r] + bb;
      }
    }
  }
}

// ---------------- BN statistics / finalize / apply ----------------
__global__ __launch_bounds__(256) void stats_kernel(const float* __restrict__ Y,
                                                    float* __restrict__ sum,
                                                    float* __restrict__ ssum) {
  const int c = threadIdx.x;                       // one channel per thread
  size_t base = (size_t)blockIdx.x * 512 * 256;    // 128 blocks x 512 rows
  float s = 0.0f, ss = 0.0f;
  for (int r = 0; r < 512; ++r) {
    float v = Y[base + (size_t)r * 256 + c];
    s += v; ss += v * v;
  }
  atomicAdd(&sum[c], s);
  atomicAdd(&ssum[c], ss);
}

__global__ __launch_bounds__(256) void finalize_bn_kernel(
    const float* __restrict__ sum, const float* __restrict__ ssum,
    const float* __restrict__ gamma, const float* __restrict__ beta,
    float* __restrict__ scale, float* __restrict__ shift) {
  const int c = threadIdx.x;
  const float inv = 1.0f / 65536.0f;               // N*H2*W2 = 16*64*64
  float m = sum[c] * inv;
  float var = ssum[c] * inv - m * m;
  float s = gamma[c] * rsqrtf(var + 1e-5f);
  scale[c] = s;
  shift[c] = beta[c] - m * s;
}

__global__ __launch_bounds__(256) void apply_bn_kernel(
    const float* __restrict__ Y1, float* __restrict__ Y2,
    unsigned short* __restrict__ X2,
    const float* __restrict__ sc1, const float* __restrict__ sh1,
    const float* __restrict__ sc2, const float* __restrict__ sh2) {
  size_t idx = (size_t)blockIdx.x * 256 + threadIdx.x;
  int c = idx & 255;
  float v1 = fmaxf(Y1[idx] * sc1[c] + sh1[c], 0.0f);   // BN11 + ReLU -> conv9 input
  X2[idx] = f2bf(v1);
  Y2[idx] = Y2[idx] * sc2[c] + sh2[c];                 // BN12 (no relu), in place
}

// ---------------- final add + relu, NHWC -> NCHW via LDS transpose ----------------
__global__ __launch_bounds__(256) void final_kernel(
    const float* __restrict__ Z, const float* __restrict__ Y2,
    const float* __restrict__ sc3, const float* __restrict__ sh3,
    float* __restrict__ out) {
  __shared__ float tile[256 * 32];
  const int b  = blockIdx.x;                 // n*128 + y*2 + xh
  const int xh = b & 1, y = (b >> 1) & 63, n = b >> 7;
  const int tid = threadIdx.x;
  const size_t base = (((size_t)n * 64 + y) * 64 + xh * 32) * 256;
  for (int i = tid; i < 32 * 256; i += 256) {
    int xr = i >> 8, c = i & 255;
    float v = Z[base + i] * sc3[c] + sh3[c] + Y2[base + i];
    tile[c * 32 + xr] = fmaxf(v, 0.0f);
  }
  __syncthreads();
  for (int i = tid; i < 32 * 256; i += 256) {
    int c = i >> 5, xr = i & 31;
    out[(((size_t)n * 256 + c) * 64 + y) * 64 + xh * 32 + xr] = tile[i];
  }
}

// ---------------- host launcher ----------------
extern "C" void kernel_launch(void* const* d_in, const int* in_sizes, int n_in,
                              void* d_out, int out_size, void* d_ws, size_t ws_size,
                              hipStream_t stream) {
  (void)in_sizes; (void)n_in; (void)out_size; (void)ws_size;
  const float* x   = (const float*)d_in[0];
  const float* w1  = (const float*)d_in[1];  const float* b1 = (const float*)d_in[2];
  const float* w2  = (const float*)d_in[3];  const float* b2 = (const float*)d_in[4];
  const float* w3  = (const float*)d_in[5];  const float* b3 = (const float*)d_in[6];
  const float* w4  = (const float*)d_in[7];  const float* b4 = (const float*)d_in[8];
  const float* w5  = (const float*)d_in[9];  const float* b5 = (const float*)d_in[10];
  const float* w6  = (const float*)d_in[11]; const float* b6 = (const float*)d_in[12];
  const float* w7  = (const float*)d_in[13]; const float* b7 = (const float*)d_in[14];
  const float* w8  = (const float*)d_in[15]; const float* b8 = (const float*)d_in[16];
  const float* w9  = (const float*)d_in[17]; const float* b9 = (const float*)d_in[18];
  const float* g11 = (const float*)d_in[19]; const float* be11 = (const float*)d_in[20];
  const float* g12 = (const float*)d_in[21]; const float* be12 = (const float*)d_in[22];
  const float* g2  = (const float*)d_in[23]; const float* be2  = (const float*)d_in[24];
  float* out = (float*)d_out;

  // workspace carve-up (256B aligned)
  char* ws = (char*)d_ws;
  size_t off = 0;
  auto carve = [&](size_t bytes) {
    char* p = ws + off;
    off += (bytes + 255) & ~(size_t)255;
    return p;
  };
  unsigned short* Xb  = (unsigned short*)carve((size_t)16 * 32 * 32 * 512 * 2);   // 16 MB
  unsigned short* Wp  = (unsigned short*)carve((size_t)2048 * 4608 * 2);          // 18 MB
  unsigned short* Wp9 = (unsigned short*)carve((size_t)256 * 2304 * 2);           // 1.1 MB
  float* Bias = (float*)carve((size_t)2048 * 4);
  float* Y1   = (float*)carve((size_t)16 * 64 * 64 * 256 * 4);                    // 64 MB
  float* Y2   = (float*)carve((size_t)16 * 64 * 64 * 256 * 4);                    // 64 MB
  unsigned short* X2 = (unsigned short*)carve((size_t)16 * 64 * 64 * 256 * 2);    // 32 MB
  float* Z    = (float*)carve((size_t)16 * 64 * 64 * 256 * 4);                    // 64 MB
  float* stats = (float*)carve((size_t)6 * 256 * 4);
  float* prm   = (float*)carve((size_t)6 * 256 * 4);
  float *s1 = stats, *q1 = stats + 256, *s2 = stats + 512, *q2 = stats + 768,
        *s3 = stats + 1024, *q3 = stats + 1280;
  float *sc1 = prm, *sh1 = prm + 256, *sc2 = prm + 512, *sh2 = prm + 768,
        *sc3 = prm + 1024, *sh3 = prm + 1280;

  // 1) packing + stat init
  pack_x_kernel<<<32768, 256, 0, stream>>>(x, Xb);
  pack_w_kernel<<<dim3(18, 2048), 256, 0, stream>>>(w1, w2, w3, w4, w5, w6, w7, w8, Wp);
  pack_w9_kernel<<<dim3(9, 256), 256, 0, stream>>>(w9, Wp9);
  pack_bias_kernel<<<8, 256, 0, stream>>>(b1, b2, b3, b4, b5, b6, b7, b8, Bias);
  zero_stats_kernel<<<1, 256, 0, stream>>>(stats);

  // 2) branch convs (both branches, all 8 sub-convs) as one implicit GEMM
  gemm1_kernel<<<dim3(64, 16), 256, 0, stream>>>(Xb, Wp, Bias, Y1, Y2);

  // 3) BN stats + finalize for both branches
  stats_kernel<<<128, 256, 0, stream>>>(Y1, s1, q1);
  stats_kernel<<<128, 256, 0, stream>>>(Y2, s2, q2);
  finalize_bn_kernel<<<1, 256, 0, stream>>>(s1, q1, g11, be11, sc1, sh1);
  finalize_bn_kernel<<<1, 256, 0, stream>>>(s2, q2, g12, be12, sc2, sh2);

  // 4) BN11+ReLU -> bf16 conv9 input; BN12 in place
  apply_bn_kernel<<<65536, 256, 0, stream>>>(Y1, Y2, X2, sc1, sh1, sc2, sh2);

  // 5) conv9 implicit GEMM
  gemm2_kernel<<<dim3(256, 2), 256, 0, stream>>>(X2, Wp9, b9, Z);

  // 6) BN2 stats + finalize
  stats_kernel<<<128, 256, 0, stream>>>(Z, s3, q3);
  finalize_bn_kernel<<<1, 256, 0, stream>>>(s3, q3, g2, be2, sc3, sh3);

  // 7) relu(bn2(Z) + Y2) -> NCHW out
  final_kernel<<<2048, 256, 0, stream>>>(Z, Y2, sc3, sh3, out);
}
